// APHetNet_18794776887890
// MI455X (gfx1250) — matrix operand order repaired and checked by reading, not compiled, not run on hardware
//
#include <hip/hip_runtime.h>
#include <hip/hip_bf16.h>

// ---------------------------------------------------------------------------
// APHetNet on MI455X (gfx1250): fully-fused bipartite GNN forward.
// Dense edge ordering (e = ue*1024+ap) -> no index arrays; BN bias
// cancellation; analytic BN1 stats from node-level Gram sums; bf16
// v_wmma_f32_16x16x32_bf16 for the E=524288-row edge MLPs.
// This rev: K-pair-packed zero-padded bf16 operands prepared once ->
// branch-free single-u32 fragment builds (no exec-mask tails, no per-block
// weight converts).
// ---------------------------------------------------------------------------

#define N_UE   512
#define N_AP   1024
#define NEDGE  (N_UE * N_AP)     // 524288
#define OUT    64
#define HIDW   128
#define EPS    1e-5f

typedef __attribute__((ext_vector_type(16))) __bf16 v16bf;
typedef __attribute__((ext_vector_type(2)))  __bf16 v2bf;
typedef __attribute__((ext_vector_type(8)))  float  v8f;

union AFrag { v16bf v; unsigned u[8]; };

__device__ __forceinline__ unsigned short f2bf(float x) {
  union { __bf16 b; unsigned short s; } c;
  c.b = (__bf16)x;                       // hardware v_cvt_bf16_f32 (RNE)
  return c.s;
}
__device__ __forceinline__ float bf2f(unsigned short h) {
  union { unsigned u; float f; } v; v.u = ((unsigned)h) << 16;
  return v.f;
}
__device__ __forceinline__ unsigned pack2(float lo, float hi) {
  union { v2bf v; unsigned u; } c;
  c.v[0] = (__bf16)lo;                   // pattern-matches v_cvt_pk_bf16_f32
  c.v[1] = (__bf16)hi;
  return c.u;
}

// ------------------------------ utility ------------------------------------
__global__ void zero_f32(float* p, int n) {
  int i = blockIdx.x * 256 + threadIdx.x;
  if (i < n) p[i] = 0.f;
}

// --- pack weight [din,ncol] f32 -> K-pair-packed bf16 [kpad2,ncol] u32 -----
// Wp[kp*ncol+n] = pack(W[2kp][n], W[2kp+1][n]), zero-padded past din.
__global__ void pack_weight(const float* __restrict__ W, int din, int ncol,
                            unsigned* __restrict__ Wp, int kpad2) {
  int idx = blockIdx.x * 256 + threadIdx.x;
  if (idx >= kpad2 * ncol) return;
  int kp = idx / ncol, n = idx % ncol;
  float lo = (2 * kp     < din) ? W[(long)(2 * kp) * ncol + n]     : 0.f;
  float hi = (2 * kp + 1 < din) ? W[(long)(2 * kp + 1) * ncol + n] : 0.f;
  Wp[idx] = pack2(lo, hi);
}

// --- pack rows [n,ld] f32 -> bf16 pairs [n,ld/2] u32 (ld even) -------------
__global__ void pack_rows(const float* __restrict__ x, long n, int ld,
                          unsigned* __restrict__ xp) {
  long idx = (long)blockIdx.x * 256 + threadIdx.x;
  int half = ld / 2;
  if (idx >= n * half) return;
  long row = idx / half; int c = (int)(idx % half);
  xp[idx] = pack2(x[row * ld + 2 * c], x[row * ld + 2 * c + 1]);
}

// --------------------- edge-attr node sums / global Gram -------------------
// glob: [cs0, cs1, g00, g01, g11]
template<bool UP>
__global__ void ea_stats_kernel(const float* __restrict__ ea,
                                float* __restrict__ sea,
                                float* __restrict__ glob) {
  const int node = blockIdx.x;
  const int cnt  = UP ? N_AP : N_UE;
  float s0 = 0, s1 = 0, g00 = 0, g01 = 0, g11 = 0;
  for (int i = threadIdx.x; i < cnt; i += blockDim.x) {
    long e = UP ? ((long)node * N_AP + i) : ((long)i * N_AP + node);
    float a = ea[e * 2 + 0], b = ea[e * 2 + 1];
    s0 += a; s1 += b; g00 += a * a; g01 += a * b; g11 += b * b;
  }
  __shared__ float r[5];
  if (threadIdx.x < 5) r[threadIdx.x] = 0.f;
  __syncthreads();
  atomicAdd(&r[0], s0); atomicAdd(&r[1], s1);
  atomicAdd(&r[2], g00); atomicAdd(&r[3], g01); atomicAdd(&r[4], g11);
  __syncthreads();
  if (threadIdx.x == 0) {
    sea[node * 2 + 0] = r[0]; sea[node * 2 + 1] = r[1];
    atomicAdd(&glob[0], r[0]); atomicAdd(&glob[1], r[1]);
    atomicAdd(&glob[2], g00 * 0.f + r[2]); atomicAdd(&glob[3], r[3]); atomicAdd(&glob[4], r[4]);
  }
}

// --------- Gram of X = [x_src[src(e)], ea(e)] over all E edges -------------
__global__ void gram_kernel(const float* __restrict__ xsrc, int nsrc, int srcLd,
                            int sc, float cmult,
                            const float* __restrict__ sea,
                            const float* __restrict__ eaglob,
                            float* __restrict__ G, float* __restrict__ cs, int din) {
  for (int i = threadIdx.x; i < din; i += blockDim.x) {
    if (i < sc) {
      float s = 0.f;
      for (int n = 0; n < nsrc; ++n) s += xsrc[(long)n * srcLd + i];
      cs[i] = s * cmult;
    } else cs[i] = eaglob[i - sc];
  }
  for (int p = threadIdx.x; p < din * din; p += blockDim.x) {
    int i = p / din, j = p % din;
    float s = 0.f;
    if (i < sc && j < sc) {
      for (int n = 0; n < nsrc; ++n)
        s += xsrc[(long)n * srcLd + i] * xsrc[(long)n * srcLd + j];
      s *= cmult;
    } else if (i < sc) {
      for (int n = 0; n < nsrc; ++n) s += xsrc[(long)n * srcLd + i] * sea[n * 2 + (j - sc)];
    } else if (j < sc) {
      for (int n = 0; n < nsrc; ++n) s += xsrc[(long)n * srcLd + j] * sea[n * 2 + (i - sc)];
    } else {
      s = eaglob[2 + (i - sc) + (j - sc)];   // g00 / g01 / g11
    }
    G[p] = s;
  }
}

// -------- BN affine from analytic stats of z = X@W (bias cancels) ----------
__global__ void bn_lin_finalize(const float* __restrict__ G, const float* __restrict__ cs,
                                const float* __restrict__ W, int wld, int din,
                                const float* __restrict__ g, const float* __restrict__ be,
                                float invE, float* __restrict__ A, float* __restrict__ B,
                                int nf) {
  int f = threadIdx.x;
  if (f >= nf) return;
  float wcol[72];
  for (int i = 0; i < din; ++i) wcol[i] = W[i * wld + f];
  float zm = 0.f;
  for (int i = 0; i < din; ++i) zm += cs[i] * wcol[i];
  zm *= invE;
  float q = 0.f;
  for (int i = 0; i < din; ++i) {
    const float* Gi = G + i * din;
    float acc = 0.f;
    for (int j = 0; j < din; ++j) acc += Gi[j] * wcol[j];
    q += wcol[i] * acc;
  }
  q *= invE;
  float var = q - zm * zm;
  float s = g[f] * rsqrtf(var + EPS);
  A[f] = s; B[f] = be[f] - s * zm;
}

// -------- BN affine from accumulated sum/sumsq (stat=[sum nf][sumsq nf]) ---
__global__ void bn_stat_finalize(const float* __restrict__ stat, float invE,
                                 const float* __restrict__ g, const float* __restrict__ be,
                                 float* __restrict__ A, float* __restrict__ B, int nf) {
  int f = threadIdx.x;
  if (f >= nf) return;
  float mu  = stat[f] * invE;
  float var = stat[nf + f] * invE - mu * mu;
  float s = g[f] * rsqrtf(var + EPS);
  A[f] = s; B[f] = be[f] - s * mu;
}

// ---------------------------------------------------------------------------
// Fused edge-message pass:  z2[e,64] = (relu(BN1(X W1))) W2   (raw, pre-BN2)
// 8 waves / WG, each wave owns a 16-edge M-tile; bf16 WMMA, f32 accum.
// All operands are pre-packed bf16 pairs -> branch-free u32 fragment builds.
// ---------------------------------------------------------------------------
template<int SC, bool UP>
__global__ __launch_bounds__(256) void edge_fused(
    const unsigned* __restrict__ srcPk,     // [nsrc, SC/2] bf16-pair u32
    const unsigned* __restrict__ eaPk,      // [NEDGE] bf16-pair u32
    const unsigned* __restrict__ W1p,       // [DINP/2, 128] bf16-pair u32
    const float* __restrict__ A1, const float* __restrict__ B1,
    const unsigned* __restrict__ W2p,       // [64, 64] bf16-pair u32
    unsigned short* __restrict__ z2,        // [NEDGE, 64] bf16
    float* __restrict__ stat2)              // [128] sum|sumsq
{
  constexpr int DIN  = SC + 2;
  constexpr int DINP = ((DIN + 31) / 32) * 32;
  constexpr int KT1  = DINP / 32;

  __shared__ unsigned short sA[8][16 * HIDW];   // 32 KB: relu(BN1(z1)) tiles
  __shared__ float sStat[128];

  const int tid  = threadIdx.x;
  const int wave = tid >> 5;
  const int lane = tid & 31;
  const int hi   = lane >> 4;
  const int ln   = lane & 15;
  if (tid < 128) sStat[tid] = 0.f;
  __syncthreads();

  const long eBase = (long)blockIdx.x * 128 + wave * 16;
  if (lane == 0) __builtin_prefetch(eaPk + eBase + 128, 0, 1);

  // ---- A fragments for GEMM1 (branch-free; tail is cndmask + zeros) -------
  AFrag a1[KT1];
  {
    const long e = eBase + ln;
    const int node = UP ? (int)(e >> 10) : (int)(e & (N_AP - 1));
    const unsigned* xr = srcPk + (long)node * (SC / 2);
    const unsigned eaw = eaPk[e];
    if constexpr (SC == 64) {
#pragma unroll
      for (int ks = 0; ks < 2; ++ks)
#pragma unroll
        for (int r = 0; r < 8; ++r)
          a1[ks].u[r] = xr[ks * 16 + r + hi * 8];
      a1[2].u[0] = hi ? 0u : eaw;       // k = 64/65 (ea) for lo half only
#pragma unroll
      for (int r = 1; r < 8; ++r) a1[2].u[r] = 0u;
    } else {                            // SC == 4, DIN = 6, DINP = 32
      a1[0].u[0] = hi ? 0u : xr[0];
      a1[0].u[1] = hi ? 0u : xr[1];
      a1[0].u[2] = hi ? 0u : eaw;
#pragma unroll
      for (int r = 3; r < 8; ++r) a1[0].u[r] = 0u;
    }
  }

  // ---- GEMM1 + BN1 affine + ReLU -> sA (bf16, A-source layout) ------------
#pragma unroll
  for (int nt = 0; nt < 8; ++nt) {
    v8f acc = {};
#pragma unroll
    for (int ks = 0; ks < KT1; ++ks) {
      AFrag b;                          // B 32x16: n = ln, kp = ks*16 + r + hi*8
      const unsigned* bp = W1p + (long)(ks * 16 + hi * 8) * HIDW + nt * 16 + ln;
#pragma unroll
      for (int r = 0; r < 8; ++r) b.u[r] = bp[(long)r * HIDW];
      acc = __builtin_amdgcn_wmma_f32_16x16x32_bf16(false, a1[ks].v, false, b.v,
                                                    (short)0, acc, false, false);
    }
    const int f = nt * 16 + ln;
    const float sc1 = A1[f], sh1 = B1[f];
#pragma unroll
    for (int r = 0; r < 8; ++r) {
      int M = r + hi * 8;
      float av = fmaxf(acc[r] * sc1 + sh1, 0.f);  // v_fma + v_max_num_f32
      sA[wave][M * HIDW + f] = f2bf(av);
    }
  }

  // ---- GEMM2: (16x128) @ (128x64) -----------------------------------------
  AFrag a2[4];
#pragma unroll
  for (int ks = 0; ks < 4; ++ks)
#pragma unroll
    for (int r = 0; r < 8; ++r) {
      int k = ks * 32 + 2 * r + hi * 16;
      a2[ks].u[r] = *(const unsigned*)&sA[wave][ln * HIDW + k];
    }

#pragma unroll
  for (int nt = 0; nt < 4; ++nt) {
    v8f acc = {};
#pragma unroll
    for (int ks = 0; ks < 4; ++ks) {
      AFrag b;
      const unsigned* bp = W2p + (long)(ks * 16 + hi * 8) * OUT + nt * 16 + ln;
#pragma unroll
      for (int r = 0; r < 8; ++r) b.u[r] = bp[(long)r * OUT];
      acc = __builtin_amdgcn_wmma_f32_16x16x32_bf16(false, a2[ks].v, false, b.v,
                                                    (short)0, acc, false, false);
    }
    const int f = nt * 16 + ln;
    float s = 0.f, ss = 0.f;
#pragma unroll
    for (int r = 0; r < 8; ++r) {
      int M = r + hi * 8;
      float v = acc[r];
      s += v; ss += v * v;
      z2[(eBase + M) * OUT + f] = f2bf(v);
    }
    atomicAdd(&sStat[f], s);
    atomicAdd(&sStat[64 + f], ss);
  }
  __syncthreads();
  if (tid < 128) atomicAdd(&stat2[tid], sStat[tid]);
}

// -------- BN2 affine + ReLU + segment-sum over edges -> agg[n_dst,64] ------
template<bool UP>
__global__ __launch_bounds__(256) void segment_reduce(
    const unsigned short* __restrict__ z2,
    const float* __restrict__ A2, const float* __restrict__ B2,
    float* __restrict__ agg) {
  const int dst   = blockIdx.x;
  const int f     = threadIdx.x & 63;
  const int strip = threadIdx.x >> 6;     // 0..3
  const int cnt   = UP ? N_UE : N_AP;
  const float a = A2[f], b = B2[f];
  float s = 0.f;
  for (int i = strip; i < cnt; i += 4) {
    long e = UP ? ((long)i * N_AP + dst) : ((long)dst * N_AP + i);
    s += fmaxf(bf2f(z2[e * OUT + f]) * a + b, 0.f);
  }
  __shared__ float red[256];
  red[threadIdx.x] = s;
  __syncthreads();
  if (strip == 0)
    agg[(long)dst * OUT + f] = red[f] + red[64 + f] + red[128 + f] + red[192 + f];
}

// -------- node update MLP (<=1024 rows): exact BN via in-LDS stats ---------
template<bool RESID>
__global__ __launch_bounds__(1024) void node_update(
    const float* __restrict__ xold, int ldOld, int dstCols, int n,
    const float* __restrict__ agg,
    const float* __restrict__ W1, const float* __restrict__ g1, const float* __restrict__ be1,
    const float* __restrict__ W2, const float* __restrict__ g2, const float* __restrict__ be2,
    float* __restrict__ xnew, float* __restrict__ z1s, float* __restrict__ z2s) {
  const int din = dstCols + OUT;
  const int row = threadIdx.x;
  __shared__ float st1[256], ab1[256], st2[120], ab2[120];
  for (int i = row; i < 256; i += blockDim.x) st1[i] = 0.f;
  for (int i = row; i < 120; i += blockDim.x) st2[i] = 0.f;
  __syncthreads();

  float fx[128];
  if (row < n) {
    for (int j = 0; j < din; ++j)
      fx[j] = (j < dstCols) ? xold[(long)row * ldOld + j]
                            : agg[(long)row * OUT + (j - dstCols)];
    for (int h = 0; h < HIDW; ++h) {
      float s = 0.f;
      for (int j = 0; j < din; ++j) s += fx[j] * W1[(long)j * HIDW + h];
      z1s[(long)row * HIDW + h] = s;
      atomicAdd(&st1[h], s);
      atomicAdd(&st1[128 + h], s * s);
    }
  }
  __syncthreads();
  if (row < 128) {
    float mu = st1[row] / n;
    float var = st1[128 + row] / n - mu * mu;
    float sc = g1[row] * rsqrtf(var + EPS);
    ab1[row] = sc; ab1[128 + row] = be1[row] - sc * mu;
  }
  __syncthreads();
  if (row < n) {
    float a[128];
    for (int h = 0; h < HIDW; ++h)
      a[h] = fmaxf(z1s[(long)row * HIDW + h] * ab1[h] + ab1[128 + h], 0.f);
    for (int o = 0; o < 60; ++o) {
      float s = 0.f;
      for (int h = 0; h < HIDW; ++h) s += a[h] * W2[(long)h * 60 + o];
      z2s[(long)row * 60 + o] = s;
      atomicAdd(&st2[o], s);
      atomicAdd(&st2[60 + o], s * s);
    }
  }
  __syncthreads();
  if (row < 60) {
    float mu = st2[row] / n;
    float var = st2[60 + row] / n - mu * mu;
    float sc = g2[row] * rsqrtf(var + EPS);
    ab2[row] = sc; ab2[60 + row] = be2[row] - sc * mu;
  }
  __syncthreads();
  if (row < n) {
    for (int j = 0; j < 4; ++j) xnew[(long)row * OUT + j] = xold[(long)row * ldOld + j];
    for (int o = 0; o < 60; ++o) {
      float v = fmaxf(z2s[(long)row * 60 + o] * ab2[o] + ab2[60 + o], 0.f);
      if (RESID) v += 0.1f * xold[(long)row * ldOld + 4 + o];
      xnew[(long)row * OUT + 4 + o] = v;
    }
  }
}

// -------- output heads: [64->4] + BN + ReLU + [4->1] + sigmoid -------------
__global__ __launch_bounds__(1024) void head_kernel(
    const float* __restrict__ x, int n,
    const float* __restrict__ W1, const float* __restrict__ b1,
    const float* __restrict__ g1, const float* __restrict__ be1,
    const float* __restrict__ W2, const float* __restrict__ b2,
    float* __restrict__ out, int outStride, int copyCols) {
  __shared__ float st[16], ab[8];
  const int row = threadIdx.x;
  if (row < 16) st[row] = 0.f;
  __syncthreads();
  float h[4];
  if (row < n) {
    for (int o = 0; o < 4; ++o) {
      float s = b1[o];
      for (int j = 0; j < OUT; ++j) s += x[(long)row * OUT + j] * W1[j * 4 + o];
      h[o] = s;
      atomicAdd(&st[o], s); atomicAdd(&st[8 + o], s * s);
    }
  }
  __syncthreads();
  if (row < 4) {
    float mu = st[row] / n;
    float var = st[8 + row] / n - mu * mu;
    float sc = g1[row] * rsqrtf(var + EPS);
    ab[row] = sc; ab[4 + row] = be1[row] - sc * mu;
  }
  __syncthreads();
  if (row < n) {
    float s = b2[0];
    for (int o = 0; o < 4; ++o) s += fmaxf(h[o] * ab[o] + ab[4 + o], 0.f) * W2[o];
    float p = 1.f / (1.f + __expf(-s));
    for (int j = 0; j < copyCols; ++j) out[(long)row * outStride + j] = x[(long)row * OUT + j];
    out[(long)row * outStride + copyCols] = p;
  }
}

// ---------------------------------------------------------------------------
extern "C" void kernel_launch(void* const* d_in, const int* in_sizes, int n_in,
                              void* d_out, int out_size, void* d_ws, size_t ws_size,
                              hipStream_t stream) {
  const float* x_ue  = (const float*)d_in[0];
  const float* x_ap  = (const float*)d_in[1];
  const float* ea_up = (const float*)d_in[4];
  const float* ea_dn = (const float*)d_in[5];

  // params flattened in dict/list insertion order: each MLP layer = W,b,g,be
  const float* P[108];
  for (int i = 0; i < 108; ++i) P[i] = (const float*)d_in[(6 + i) < n_in ? (6 + i) : (n_in - 1)];

  // workspace carve-up
  char* w = (char*)d_ws; size_t off = 0;
  auto alloc = [&](size_t bytes) -> void* {
    void* p = w + off; off = (off + bytes + 255) & ~(size_t)255; return p;
  };
  unsigned short* z2 = (unsigned short*)alloc((size_t)NEDGE * OUT * 2);   // 67 MB
  unsigned* eaPkU = (unsigned*)alloc((size_t)NEDGE * 4);                  // 2 MB
  unsigned* eaPkD = (unsigned*)alloc((size_t)NEDGE * 4);                  // 2 MB
  unsigned* srcPk = (unsigned*)alloc((size_t)N_AP * 32 * 4);              // 128 KB
  unsigned* W1p   = (unsigned*)alloc((size_t)48 * HIDW * 4);
  unsigned* W2p   = (unsigned*)alloc((size_t)64 * OUT * 4);
  float* agg  = (float*)alloc((size_t)N_AP * OUT * 4);
  float* xue0 = (float*)alloc((size_t)N_UE * OUT * 4);
  float* xue1 = (float*)alloc((size_t)N_UE * OUT * 4);
  float* xap0 = (float*)alloc((size_t)N_AP * OUT * 4);
  float* xap1 = (float*)alloc((size_t)N_AP * OUT * 4);
  float* seaU = (float*)alloc((size_t)N_UE * 2 * 4);
  float* seaD = (float*)alloc((size_t)N_AP * 2 * 4);
  float* eagU = (float*)alloc(8 * 4);
  float* eagD = (float*)alloc(8 * 4);
  float* G    = (float*)alloc(66 * 66 * 4);
  float* cs   = (float*)alloc(72 * 4);
  float* bnA1 = (float*)alloc(128 * 4);
  float* bnB1 = (float*)alloc(128 * 4);
  float* bnA2 = (float*)alloc(64 * 4);
  float* bnB2 = (float*)alloc(64 * 4);
  float* stat2 = (float*)alloc(128 * 4);
  float* z1n  = (float*)alloc((size_t)N_AP * HIDW * 4);
  float* z2n  = (float*)alloc((size_t)N_AP * 64 * 4);
  const float invE = 1.f / (float)NEDGE;

  // one-time per-direction prep: ea bf16 pair-pack + ea stats
  pack_rows<<<(NEDGE + 255) / 256, 256, 0, stream>>>(ea_up, NEDGE, 2, eaPkU);
  pack_rows<<<(NEDGE + 255) / 256, 256, 0, stream>>>(ea_dn, NEDGE, 2, eaPkD);
  zero_f32<<<1, 256, 0, stream>>>(eagU, 8);
  zero_f32<<<1, 256, 0, stream>>>(eagD, 8);
  ea_stats_kernel<true><<<N_UE, 256, 0, stream>>>(ea_up, seaU, eagU);
  ea_stats_kernel<false><<<N_AP, 256, 0, stream>>>(ea_dn, seaD, eagD);

  struct Pass {
    bool up; int sc; const float* src; int srcLd;
    const float* ea; const unsigned* eaPk; const float* sea; const float* eag; float cmult;
    int msg, upd;                    // base indices into P
    const float* xold; int ldOld; int dstCols; int nDst; bool resid;
    float* xnew;
  };
  Pass ps[6] = {
    { true,  4,  x_ue, 4,  ea_up, eaPkU, seaU, eagU, (float)N_AP,  0,  8, x_ap, 4, 4, N_AP, false, xap0 },
    { false, 64, xap0, 64, ea_dn, eaPkD, seaD, eagD, (float)N_UE, 16, 24, x_ue, 4, 4, N_UE, false, xue0 },
    { true,  64, xue0, 64, ea_up, eaPkU, seaU, eagU, (float)N_AP, 32, 40, xap0, 64, 64, N_AP, true,  xap1 },
    { false, 64, xap1, 64, ea_dn, eaPkD, seaD, eagD, (float)N_UE, 48, 56, xue0, 64, 64, N_UE, true,  xue1 },
    { true,  64, xue1, 64, ea_up, eaPkU, seaU, eagU, (float)N_AP, 64, 72, xap1, 64, 64, N_AP, true,  xap0 },
    { false, 64, xap0, 64, ea_dn, eaPkD, seaD, eagD, (float)N_UE, 80, 88, xue1, 64, 64, N_UE, true,  xue0 },
  };

  for (int p = 0; p < 6; ++p) {
    const Pass& q = ps[p];
    const int nsrc = q.up ? N_UE : N_AP;
    const int kpad2 = (q.sc == 4) ? 16 : 48;     // DINP/2
    const float *mW1 = P[q.msg], *mG1 = P[q.msg + 2], *mB1e = P[q.msg + 3];
    const float *mW2 = P[q.msg + 4], *mG2 = P[q.msg + 6], *mB2e = P[q.msg + 7];

    // per-pass operand prep (tiny)
    pack_rows<<<(nsrc * q.sc / 2 + 255) / 256, 256, 0, stream>>>(q.src, nsrc, q.sc, srcPk);
    pack_weight<<<(kpad2 * HIDW + 255) / 256, 256, 0, stream>>>(mW1, q.sc + 2, HIDW, W1p, kpad2);
    pack_weight<<<(64 * OUT + 255) / 256, 256, 0, stream>>>(mW2, HIDW, OUT, W2p, 64);

    gram_kernel<<<1, 256, 0, stream>>>(q.src, nsrc, q.srcLd, q.sc, q.cmult,
                                       q.sea, q.eag, G, cs, q.sc + 2);
    bn_lin_finalize<<<1, 128, 0, stream>>>(G, cs, mW1, HIDW, q.sc + 2,
                                           mG1, mB1e, invE, bnA1, bnB1, 128);
    zero_f32<<<1, 256, 0, stream>>>(stat2, 128);
    if (q.sc == 4) {
      edge_fused<4, true><<<NEDGE / 128, 256, 0, stream>>>(
          srcPk, q.eaPk, W1p, bnA1, bnB1, W2p, z2, stat2);
    } else if (q.up) {
      edge_fused<64, true><<<NEDGE / 128, 256, 0, stream>>>(
          srcPk, q.eaPk, W1p, bnA1, bnB1, W2p, z2, stat2);
    } else {
      edge_fused<64, false><<<NEDGE / 128, 256, 0, stream>>>(
          srcPk, q.eaPk, W1p, bnA1, bnB1, W2p, z2, stat2);
    }
    bn_stat_finalize<<<1, 64, 0, stream>>>(stat2, invE, mG2, mB2e, bnA2, bnB2, 64);
    if (q.up) segment_reduce<true><<<N_AP, 256, 0, stream>>>(z2, bnA2, bnB2, agg);
    else      segment_reduce<false><<<N_UE, 256, 0, stream>>>(z2, bnA2, bnB2, agg);

    const float *uW1 = P[q.upd], *uG1 = P[q.upd + 2], *uB1e = P[q.upd + 3];
    const float *uW2 = P[q.upd + 4], *uG2 = P[q.upd + 6], *uB2e = P[q.upd + 7];
    if (q.resid)
      node_update<true><<<1, 1024, 0, stream>>>(q.xold, q.ldOld, q.dstCols, q.nDst,
          agg, uW1, uG1, uB1e, uW2, uG2, uB2e, q.xnew, z1n, z2n);
    else
      node_update<false><<<1, 1024, 0, stream>>>(q.xold, q.ldOld, q.dstCols, q.nDst,
          agg, uW1, uG1, uB1e, uW2, uG2, uB2e, q.xnew, z1n, z2n);
  }

  // heads: x_ue -> [512,5] (4 copied cols + power), x_ap -> [1024,1]
  float* outp = (float*)d_out;
  head_kernel<<<1, 1024, 0, stream>>>(xue0, N_UE,
      P[96], P[97], P[98], P[99], P[100], P[101], outp, 5, 4);
  head_kernel<<<1, 1024, 0, stream>>>(xap0, N_AP,
      P[102], P[103], P[104], P[105], P[106], P[107], outp + (size_t)N_UE * 5, 1, 0);
}
